// PairwiseScorer_12867722019310
// MI455X (gfx1250) — matrix operand order repaired and checked by reading, not compiled, not run on hardware
//
#include <hip/hip_runtime.h>

#define G_DIM 620
#define D_DIM 1920       // 3*620 + 60
#define H_DIM 150
#define NT 10            // N tiles of 16 (150 padded to 160)
#define KSTEPS 60        // 1920 / 32
#define MROWS 64         // pairs per block (4 M-tiles of 16)
#define ROWU  40         // padded row stride within a ks-block, in ushorts (80 B)
#define KSU   (MROWS * ROWU)   // ks-block stride in ushorts (2560 -> 5120 B)
#define KMAX 160
#define ROWLEN (KMAX + 1)

typedef __attribute__((ext_vector_type(16))) __bf16 v16bf;
typedef __attribute__((ext_vector_type(8)))  float  v8f;
typedef __attribute__((ext_vector_type(4)))  unsigned int v4u;

union FragU { v16bf v; v4u u4[2]; };

__device__ __forceinline__ unsigned short f2bf(float x) {
    unsigned int u = __float_as_uint(x);
    u += 0x7FFFu + ((u >> 16) & 1u);   // round-to-nearest-even
    return (unsigned short)(u >> 16);
}

__device__ __forceinline__ unsigned long long pack4bf(float a, float b, float c, float d) {
    unsigned long long r =  (unsigned long long)f2bf(a)
                         | ((unsigned long long)f2bf(b) << 16)
                         | ((unsigned long long)f2bf(c) << 32)
                         | ((unsigned long long)f2bf(d) << 48);
    return r;
}

// Repack W1 [1920 x 150] f32 row-major into bf16 WMMA B-fragments:
// layout [nt][ks][lane][j], lane holds column n = nt*16 + (lane&15),
// K values k = ks*32 + (lane>>4)*16 + j  (j = 0..15).
__global__ void __launch_bounds__(256)
prep_w1_frag(const float* __restrict__ W1, unsigned short* __restrict__ w1f) {
    int idx = blockIdx.x * blockDim.x + threadIdx.x;
    if (idx >= NT * KSTEPS * 32 * 16) return;
    int j    = idx & 15;
    int lane = (idx >> 4) & 31;
    int ks   = (idx >> 9) % KSTEPS;
    int nt   = idx / (512 * KSTEPS);
    int n = nt * 16 + (lane & 15);
    int k = ks * 32 + ((lane >> 4) << 4) + j;
    float v = (n < H_DIM) ? W1[k * H_DIM + n] : 0.0f;
    w1f[idx] = f2bf(v);
}

// One block = 64 pairs; 10 waves, one N-tile each with 4 accumulators.
// A tile stored ks-major in LDS: [ks][row][32 K] with 80B row pitch so all
// 8 A-fragment loads per K-step share one base address (imm16 offsets only).
__global__ void __launch_bounds__(320)
pair_gemm(const float* __restrict__ g_i, const float* __restrict__ msc,
          const float* __restrict__ demb, const float* __restrict__ gemb,
          const float* __restrict__ semb,
          const unsigned short* __restrict__ w1f,
          const float* __restrict__ b1, const float* __restrict__ W2,
          const float* __restrict__ b2,
          const int* __restrict__ mids, const int* __restrict__ aids,
          const int* __restrict__ dists, const int* __restrict__ genres,
          const int* __restrict__ spks,
          float* __restrict__ coref, int P)
{
    __shared__ __align__(16) union {
        unsigned short a[KSTEPS * KSU];         // bf16 A tile, ks-major (307200 B)
        float h[MROWS * NT * 16];               // reused after K-loop (40960 B)
    } sm;
    __shared__ float w2s[NT * 16];
    __shared__ float b1s[NT * 16];

    const int p0  = blockIdx.x * MROWS;
    const int tid = threadIdx.x;

    if (tid < NT * 16) {
        w2s[tid] = (tid < H_DIM) ? W2[tid] : 0.0f;
        b1s[tid] = (tid < H_DIM) ? b1[tid] : 0.0f;
    }

    // ---- build A tile in LDS, vectorized float4; d = 4*s (620 = 155*4) ----
    const float4* g4 = (const float4*)g_i;      // 155 float4 per row
    for (int idx = tid; idx < MROWS * 480; idx += 320) {
        int r = idx / 480;
        int s = idx - r * 480;
        int p = p0 + r;
        float x0 = 0.f, x1 = 0.f, x2 = 0.f, x3 = 0.f;
        if (p < P) {
            int mi = mids[p];
            int ai = aids[p];
            if (s < 310) {
                float4 v = g4[(s < 155 ? mi : ai) * 155 + (s < 155 ? s : s - 155)];
                x0 = v.x; x1 = v.y; x2 = v.z; x3 = v.w;
            } else if (s < 465) {
                float4 u = g4[mi * 155 + (s - 310)];
                float4 v = g4[ai * 155 + (s - 310)];
                x0 = u.x * v.x; x1 = u.y * v.y; x2 = u.z * v.z; x3 = u.w * v.w;
            } else {
                int j = (s - 465) * 4;          // 0..56, groups of 4 never straddle 20/40
                const float* src;
                if (j < 20) {
                    int ds = dists[p];
                    int b = (ds >= 1) + (ds >= 2) + (ds >= 3) + (ds >= 4) +
                            (ds >= 8) + (ds >= 16) + (ds >= 32) + (ds >= 64);
                    src = demb + b * 20 + j;
                } else if (j < 40) {
                    src = gemb + genres[p] * 20 + (j - 20);
                } else {
                    src = semb + spks[p] * 20 + (j - 40);
                }
                x0 = src[0]; x1 = src[1]; x2 = src[2]; x3 = src[3];
            }
        }
        // d = 4*s = ks*32 + kin, ks = s>>3, kin = 4*(s&7)
        int ks  = s >> 3;
        int kin = (s & 7) << 2;
        *(unsigned long long*)&sm.a[ks * KSU + r * ROWU + kin] = pack4bf(x0, x1, x2, x3);
    }
    __syncthreads();

    const int wave = tid >> 5;   // == N-tile index, 0..9
    const int lane = tid & 31;

    // Per K-step base: lane row m = lane&15, chunk (lane>>4)*16; fragment rows
    // m, m+16, m+32, m+48 at immediates r*ROWU (+8 for second half of K chunk).
    const unsigned short* abase =
        &sm.a[(lane & 15) * ROWU + ((lane >> 4) << 4)];
    const unsigned short* pb = w1f + ((size_t)(wave * KSTEPS) * 32 + lane) * 16;

    v8f acc0 = {}, acc1 = {}, acc2 = {}, acc3 = {};
    #pragma unroll 2
    for (int ks = 0; ks < KSTEPS; ++ks) {
        FragU A0, A1, A2, A3, B;
        B.u4[0]  = *(const v4u*)(pb);
        B.u4[1]  = *(const v4u*)(pb + 8);
        A0.u4[0] = *(const v4u*)(abase);
        A0.u4[1] = *(const v4u*)(abase + 8);
        A1.u4[0] = *(const v4u*)(abase + 16 * ROWU);
        A1.u4[1] = *(const v4u*)(abase + 16 * ROWU + 8);
        A2.u4[0] = *(const v4u*)(abase + 32 * ROWU);
        A2.u4[1] = *(const v4u*)(abase + 32 * ROWU + 8);
        A3.u4[0] = *(const v4u*)(abase + 48 * ROWU);
        A3.u4[1] = *(const v4u*)(abase + 48 * ROWU + 8);
        acc0 = __builtin_amdgcn_wmma_f32_16x16x32_bf16(
                   false, A0.v, false, B.v, (short)0, acc0, false, false);
        acc1 = __builtin_amdgcn_wmma_f32_16x16x32_bf16(
                   false, A1.v, false, B.v, (short)0, acc1, false, false);
        acc2 = __builtin_amdgcn_wmma_f32_16x16x32_bf16(
                   false, A2.v, false, B.v, (short)0, acc2, false, false);
        acc3 = __builtin_amdgcn_wmma_f32_16x16x32_bf16(
                   false, A3.v, false, B.v, (short)0, acc3, false, false);
        abase += KSU;
        pb    += 512;
    }
    __syncthreads();   // everyone done reading A before LDS reuse

    // ---- h = relu(acc + b1) into LDS (C/D layout: VGPR v, lane -> m,n) ----
    {
        int n = wave * 16 + (lane & 15);
        float bn = b1s[n];
        #pragma unroll
        for (int v = 0; v < 8; ++v) {
            int m = v + ((lane >> 4) << 3);
            float h0 = acc0[v] + bn;
            float h1 = acc1[v] + bn;
            float h2 = acc2[v] + bn;
            float h3 = acc3[v] + bn;
            sm.h[m * (NT * 16) + n]        = h0 > 0.0f ? h0 : 0.0f;
            sm.h[(m + 16) * (NT * 16) + n] = h1 > 0.0f ? h1 : 0.0f;
            sm.h[(m + 32) * (NT * 16) + n] = h2 > 0.0f ? h2 : 0.0f;
            sm.h[(m + 48) * (NT * 16) + n] = h3 > 0.0f ? h3 : 0.0f;
        }
    }
    __syncthreads();

    // ---- s_ij = h . W2 + b2 ; combine with mention scores ----
    if (tid < MROWS) {
        int p = p0 + tid;
        if (p < P) {
            float s = b2[0];
            #pragma unroll 5
            for (int n = 0; n < H_DIM; ++n)
                s += sm.h[tid * (NT * 16) + n] * w2s[n];
            int mi = mids[p], ai = aids[p];
            coref[p] = s + msc[mi] + msc[ai];
        }
    }
}

// One block per output row. Row 0 is the special epsilon-only row;
// row m+1 handles mention m via binary search on sorted mention_ids.
__global__ void __launch_bounds__(256)
seg_softmax(const float* __restrict__ coref, const int* __restrict__ mids,
            int P, float* __restrict__ out)
{
    __shared__ float red[256];
    __shared__ int sStart, sEnd;
    const int row = blockIdx.x;
    float* orow = out + (size_t)row * ROWLEN;

    if (row == 0) {
        for (int i = threadIdx.x; i < ROWLEN; i += 256)
            orow[i] = (i == 0) ? 1.0f : 1000.0f;
        return;
    }
    const int m = row - 1;
    if (threadIdx.x == 0) {
        int lo = 0, hi = P;
        while (lo < hi) { int mid = (lo + hi) >> 1; if (mids[mid] < m) lo = mid + 1; else hi = mid; }
        sStart = lo;
        hi = P;
        while (lo < hi) { int mid = (lo + hi) >> 1; if (mids[mid] < m + 1) lo = mid + 1; else hi = mid; }
        sEnd = lo;
    }
    __syncthreads();
    const int start = sStart, end = sEnd, cnt = end - start;

    // max (epsilon logit 0.0 always in the group)
    float mx = 0.0f;
    for (int i = start + threadIdx.x; i < end; i += 256) mx = fmaxf(mx, coref[i]);
    red[threadIdx.x] = mx; __syncthreads();
    for (int s = 128; s > 0; s >>= 1) {
        if (threadIdx.x < s) red[threadIdx.x] = fmaxf(red[threadIdx.x], red[threadIdx.x + s]);
        __syncthreads();
    }
    const float segmax = red[0];
    __syncthreads();

    // sum of exp
    float sum = 0.0f;
    for (int i = start + threadIdx.x; i < end; i += 256) sum += __expf(coref[i] - segmax);
    red[threadIdx.x] = sum; __syncthreads();
    for (int s = 128; s > 0; s >>= 1) {
        if (threadIdx.x < s) red[threadIdx.x] += red[threadIdx.x + s];
        __syncthreads();
    }
    const float eps = __expf(-segmax);
    const float inv = 1.0f / (red[0] + eps);

    for (int i = threadIdx.x; i < ROWLEN; i += 256) {
        float v;
        if (i < cnt)       v = __expf(coref[start + i] - segmax) * inv;
        else if (i == cnt) v = eps * inv;
        else               v = 1000.0f;
        orow[i] = v;
    }
}

extern "C" void kernel_launch(void* const* d_in, const int* in_sizes, int n_in,
                              void* d_out, int out_size, void* d_ws, size_t ws_size,
                              hipStream_t stream)
{
    (void)n_in; (void)out_size; (void)ws_size;
    const float* g_i  = (const float*)d_in[0];
    const float* msc  = (const float*)d_in[1];
    const float* demb = (const float*)d_in[2];
    const float* gemb = (const float*)d_in[3];
    const float* semb = (const float*)d_in[4];
    const float* W1   = (const float*)d_in[5];
    const float* b1   = (const float*)d_in[6];
    const float* W2   = (const float*)d_in[7];
    const float* b2   = (const float*)d_in[8];
    const int* mids   = (const int*)d_in[9];
    const int* aids   = (const int*)d_in[10];
    const int* dists  = (const int*)d_in[11];
    const int* genres = (const int*)d_in[12];
    const int* spks   = (const int*)d_in[13];
    const int P = in_sizes[9];
    const int M = in_sizes[1];

    unsigned short* w1f = (unsigned short*)d_ws;
    float* coref = (float*)((char*)d_ws +
                   (size_t)(NT * KSTEPS * 32 * 16) * sizeof(unsigned short));

    const int fragElems = NT * KSTEPS * 32 * 16;
    prep_w1_frag<<<(fragElems + 255) / 256, 256, 0, stream>>>(W1, w1f);

    pair_gemm<<<(P + MROWS - 1) / MROWS, 320, 0, stream>>>(
        g_i, msc, demb, gemb, semb, w1f, b1, W2, b2,
        mids, aids, dists, genres, spks, coref, P);

    seg_softmax<<<M + 1, 256, 0, stream>>>(coref, mids, P, (float*)d_out);
}
